// SANet_61065845015013
// MI455X (gfx1250) — matrix operand order repaired and checked by reading, not compiled, not run on hardware
//
#include <hip/hip_runtime.h>
#include <hip/hip_bf16.h>
#include <math.h>

// Problem constants (reference: content/style [4,512,64,64] fp32)
#define BATCH 4
#define CH    512
#define HWSZ  4096   // 64*64

typedef __bf16 bf16_t;
typedef __attribute__((ext_vector_type(16))) __bf16        v16bf;
typedef __attribute__((ext_vector_type(8)))  float         v8f;
typedef __attribute__((ext_vector_type(4)))  unsigned int  u32x4;

union BF16Tile { v16bf v; u32x4 q[2]; };

// ---------------------------------------------------------------------------
// WMMA operand loaders (bf16, 16x16x32, wave32).
// A (16x32, MxK): lane L holds row M=L%16; K = {base..base+7, base+16..base+23},
//   base = (L>=16 ? 8 : 0)  -> two contiguous 16B loads.
// B (32x16, KxN): lane L holds col N=L%16; K = {base..base+7, base+16..base+23},
//   base = (L>=16 ? 8 : 0) -> two contiguous 16B loads (B element (K,N) read from row-major [N][K]).
// ---------------------------------------------------------------------------
static __device__ inline v16bf load_tile_A(const bf16_t* __restrict__ base,
                                           int ld, int row0, int kb, int lane) {
  const int r = row0 + (lane & 15);
  const int k = kb + ((lane & 16) ? 8 : 0);
  const bf16_t* p = base + (size_t)r * ld + k;
  BF16Tile t;
  t.q[0] = *reinterpret_cast<const u32x4*>(p);
  t.q[1] = *reinterpret_cast<const u32x4*>(p + 16);
  return t.v;
}

static __device__ inline v16bf load_tile_B(const bf16_t* __restrict__ base,
                                           int ld, int col0, int kb, int lane) {
  const int n = col0 + (lane & 15);
  const int k = kb + ((lane & 16) ? 8 : 0);
  const bf16_t* p = base + (size_t)n * ld + k;
  BF16Tile t;
  t.q[0] = *reinterpret_cast<const u32x4*>(p);
  t.q[1] = *reinterpret_cast<const u32x4*>(p + 16);
  return t.v;
}

// ---------------------------------------------------------------------------
// Per-(b,c) mean + 1/sqrt(unbiased var + eps) over HW spatial elements.
// ---------------------------------------------------------------------------
__global__ __launch_bounds__(256) void stats_kernel(const float* __restrict__ x,
                                                    float* __restrict__ mean,
                                                    float* __restrict__ rstd) {
  const int row = blockIdx.x;  // b*CH + c
  const float* p = x + (size_t)row * HWSZ;
  float s1 = 0.0f, s2 = 0.0f;
  for (int i = threadIdx.x; i < HWSZ; i += 256) {
    float v = p[i];
    s1 += v; s2 += v * v;
  }
  __shared__ float sh1[256], sh2[256];
  sh1[threadIdx.x] = s1; sh2[threadIdx.x] = s2;
  __syncthreads();
  for (int ofs = 128; ofs > 0; ofs >>= 1) {
    if (threadIdx.x < ofs) {
      sh1[threadIdx.x] += sh1[threadIdx.x + ofs];
      sh2[threadIdx.x] += sh2[threadIdx.x + ofs];
    }
    __syncthreads();
  }
  if (threadIdx.x == 0) {
    const float n = (float)HWSZ;
    float m   = sh1[0] / n;
    float var = (sh2[0] - n * m * m) / (n - 1.0f) + 1e-5f;
    mean[row] = m;
    rstd[row] = rsqrtf(var);
  }
}

// ---------------------------------------------------------------------------
// fp32 [C,HW] -> bf16 [HW,C] transpose, optional mean/var normalization.
// ---------------------------------------------------------------------------
__global__ __launch_bounds__(256) void transnorm_kernel(const float* __restrict__ src,
                                                        bf16_t* __restrict__ dst,
                                                        const float* __restrict__ mean,
                                                        const float* __restrict__ rstd,
                                                        int useNorm) {
  const int b  = blockIdx.z;
  const int c0 = blockIdx.y * 32;
  const int n0 = blockIdx.x * 32;
  const int tx = threadIdx.x, ty = threadIdx.y;  // 32 x 8
  __shared__ float tile[32][33];
  const float* s = src + (size_t)b * CH * HWSZ;
  #pragma unroll
  for (int i = 0; i < 32; i += 8) {
    const int c = c0 + ty + i;
    float v = s[(size_t)c * HWSZ + n0 + tx];
    if (useNorm) v = (v - mean[b * CH + c]) * rstd[b * CH + c];
    tile[ty + i][tx] = v;
  }
  __syncthreads();
  bf16_t* d = dst + (size_t)b * HWSZ * CH;
  #pragma unroll
  for (int i = 0; i < 32; i += 8) {
    const int n = n0 + ty + i;
    d[(size_t)n * CH + c0 + tx] = (bf16_t)tile[tx][ty + i];
  }
}

__global__ void f32_to_bf16_kernel(const float* __restrict__ src,
                                   bf16_t* __restrict__ dst, int n) {
  int i = blockIdx.x * blockDim.x + threadIdx.x;
  if (i < n) dst[i] = (bf16_t)src[i];
}

// ---------------------------------------------------------------------------
// NT-GEMM: D[m,n] = sum_k Arow[m][k] * Brow[n][k]  (bf16 in, f32 accum).
// Block = 8 waves, 128x64 tile; each wave computes 32x32 (2x2 WMMA D tiles).
// BIAS_MODE: 0 none, 1 per-col (N), 2 per-row (M). ADD_SRC adds residual.
// ---------------------------------------------------------------------------
template<int BIAS_MODE, bool ADD_SRC, bool OUT_BF16>
__global__ __launch_bounds__(256) void gemm_nt_kernel(
    const bf16_t* __restrict__ A, int lda, long long sA,
    const bf16_t* __restrict__ B, int ldb, long long sB,
    const float*  __restrict__ bias,
    const float*  __restrict__ add, long long sAdd,
    void* __restrict__ out, int ldd, long long sO,
    int K) {
  const int b    = blockIdx.z;
  const int lane = threadIdx.x & 31;
  const int wid  = threadIdx.x >> 5;
  const int mt   = blockIdx.y * 64 + (wid >> 1) * 32;
  const int nt   = blockIdx.x * 64 + (wid & 1) * 32;
  const bf16_t* Ab = A + (size_t)b * sA;
  const bf16_t* Bb = B + (size_t)b * sB;

  v8f zero = {};
  v8f acc[2][2];
  acc[0][0] = zero; acc[0][1] = zero; acc[1][0] = zero; acc[1][1] = zero;

  for (int kb = 0; kb < K; kb += 32) {
    v16bf a0 = load_tile_A(Ab, lda, mt,      kb, lane);
    v16bf a1 = load_tile_A(Ab, lda, mt + 16, kb, lane);
    v16bf b0 = load_tile_B(Bb, ldb, nt,      kb, lane);
    v16bf b1 = load_tile_B(Bb, ldb, nt + 16, kb, lane);
    acc[0][0] = __builtin_amdgcn_wmma_f32_16x16x32_bf16(false, a0, false, b0, (short)0, acc[0][0], false, false);
    acc[0][1] = __builtin_amdgcn_wmma_f32_16x16x32_bf16(false, a0, false, b1, (short)0, acc[0][1], false, false);
    acc[1][0] = __builtin_amdgcn_wmma_f32_16x16x32_bf16(false, a1, false, b0, (short)0, acc[1][0], false, false);
    acc[1][1] = __builtin_amdgcn_wmma_f32_16x16x32_bf16(false, a1, false, b1, (short)0, acc[1][1], false, false);
  }

  const int half8 = (lane & 16) ? 8 : 0;
  #pragma unroll
  for (int i = 0; i < 2; ++i) {
    #pragma unroll
    for (int j = 0; j < 2; ++j) {
      const int col = nt + j * 16 + (lane & 15);
      const float bc = (BIAS_MODE == 1) ? bias[col] : 0.0f;
      #pragma unroll
      for (int r = 0; r < 8; ++r) {
        const int row = mt + i * 16 + r + half8;
        float v = acc[i][j][r];
        if (BIAS_MODE == 1) v += bc;
        if (BIAS_MODE == 2) v += bias[row];
        if (ADD_SRC) v += add[(size_t)b * sAdd + (size_t)row * ldd + col];
        const size_t idx = (size_t)b * sO + (size_t)row * ldd + col;
        if (OUT_BF16) ((bf16_t*)out)[idx] = (bf16_t)v;
        else          ((float*)out)[idx]  = v;
      }
    }
  }
}

// ---------------------------------------------------------------------------
// Fused flash attention: S = F^T G (never materialized), online softmax,
// O^T = P H^T.  Block = 8 waves, owns 16 query rows (n), loops 64-key chunks.
// FT,GT: [HW,C] bf16 rows; H: [C,HW] bf16 rows; OT out: [HW,C] bf16.
// Wave w computes the 16x16 S tile for its 16 keys, then accumulates 128
// output channels (8 f32 D tiles) with P shared through LDS.
// ---------------------------------------------------------------------------
__global__ __launch_bounds__(256) void flash_attn_kernel(
    const bf16_t* __restrict__ FT, const bf16_t* __restrict__ GT,
    const bf16_t* __restrict__ Hm, bf16_t* __restrict__ OT) {
  const int b     = blockIdx.y;
  const int n0    = blockIdx.x * 16;
  const int tid   = threadIdx.x;
  const int lane  = tid & 31;
  const int wid   = tid >> 5;
  const int half8 = (lane & 16) ? 8 : 0;

  const bf16_t* Fb = FT + (size_t)b * HWSZ * CH;
  const bf16_t* Gb = GT + (size_t)b * HWSZ * CH;
  const bf16_t* Hb = Hm + (size_t)b * CH * HWSZ;
  bf16_t*       Ob = OT + (size_t)b * HWSZ * CH;

  __shared__ bf16_t Fsh[16 * CH];   // 16 query rows of F^T (16 KB)
  __shared__ bf16_t Psh[16 * 64];   // shared P tile (2 KB)
  __shared__ float  m_run[16], l_run[16], alpha_sh[16];
  __shared__ float  wmax[4][16], wsum[4][16];

  {  // preload F rows (16*512 bf16 = 1024 x 16B)
    const u32x4* src = reinterpret_cast<const u32x4*>(Fb + (size_t)n0 * CH);
    u32x4* dst = reinterpret_cast<u32x4*>(Fsh);
    for (int i = tid; i < 16 * CH / 8; i += 256) dst[i] = src[i];
  }
  if (tid < 16) { m_run[tid] = -INFINITY; l_run[tid] = 0.0f; }
  __syncthreads();

  v8f zero = {};
  v8f oacc[8];
  #pragma unroll
  for (int t = 0; t < 8; ++t) oacc[t] = zero;
  const int c0 = wid * 128;  // this wave's output-channel range

  for (int mb = 0; mb < HWSZ; mb += 64) {
    const int mw = mb + wid * 16;

    // ---- S tile: 16 queries x 16 keys, K-loop over C=512 ----
    v8f s = zero;
    for (int kb = 0; kb < CH; kb += 32) {
      v16bf af = load_tile_A(Fsh, CH, 0, kb, lane);
      v16bf bg = load_tile_B(Gb, CH, mw, kb, lane);
      s = __builtin_amdgcn_wmma_f32_16x16x32_bf16(false, af, false, bg, (short)0, s, false, false);
    }

    // ---- per-row max within this wave's 16 keys (16-lane halves) ----
    float rmax[8];
    #pragma unroll
    for (int r = 0; r < 8; ++r) {
      float v = s[r];
      v = fmaxf(v, __shfl_xor(v, 1, 32));
      v = fmaxf(v, __shfl_xor(v, 2, 32));
      v = fmaxf(v, __shfl_xor(v, 4, 32));
      v = fmaxf(v, __shfl_xor(v, 8, 32));
      rmax[r] = v;
    }
    if ((lane & 15) == 0) {
      #pragma unroll
      for (int r = 0; r < 8; ++r) wmax[wid][r + half8] = rmax[r];
    }
    __syncthreads();

    // ---- combine running max, compute rescale alpha ----
    if (tid < 16) {
      float mo = m_run[tid];
      float mn = fmaxf(fmaxf(fmaxf(wmax[0][tid], wmax[1][tid]),
                             fmaxf(wmax[2][tid], wmax[3][tid])), mo);
      m_run[tid]    = mn;
      alpha_sh[tid] = __expf(mo - mn);  // first iter: exp(-inf - mn) = 0
    }
    __syncthreads();

    // ---- P = exp(S - m), row sums, publish P tile (bf16) to LDS ----
    float mrow[8], pv[8];
    #pragma unroll
    for (int r = 0; r < 8; ++r) mrow[r] = m_run[r + half8];
    #pragma unroll
    for (int r = 0; r < 8; ++r) pv[r] = __expf(s[r] - mrow[r]);
    #pragma unroll
    for (int r = 0; r < 8; ++r) {
      float v = pv[r];
      v += __shfl_xor(v, 1, 32);
      v += __shfl_xor(v, 2, 32);
      v += __shfl_xor(v, 4, 32);
      v += __shfl_xor(v, 8, 32);
      if ((lane & 15) == 0) wsum[wid][r + half8] = v;
      Psh[(r + half8) * 64 + wid * 16 + (lane & 15)] = (bf16_t)pv[r];
    }
    __syncthreads();

    if (tid < 16) {
      l_run[tid] = l_run[tid] * alpha_sh[tid]
                 + wsum[0][tid] + wsum[1][tid] + wsum[2][tid] + wsum[3][tid];
    }

    // ---- rescale O accumulators by alpha, then O += P x H^T ----
    float arow[8];
    #pragma unroll
    for (int r = 0; r < 8; ++r) arow[r] = alpha_sh[r + half8];
    #pragma unroll
    for (int t = 0; t < 8; ++t) {
      #pragma unroll
      for (int r = 0; r < 8; ++r) oacc[t][r] *= arow[r];
    }
    #pragma unroll
    for (int kk = 0; kk < 2; ++kk) {
      v16bf pa = load_tile_A(Psh, 64, 0, kk * 32, lane);
      #pragma unroll
      for (int t = 0; t < 8; ++t) {
        v16bf hb = load_tile_B(Hb, HWSZ, c0 + t * 16, mb + kk * 32, lane);
        oacc[t] = __builtin_amdgcn_wmma_f32_16x16x32_bf16(false, pa, false, hb, (short)0, oacc[t], false, false);
      }
    }
    __syncthreads();  // protect Psh/wmax/alpha before next chunk
  }

  // ---- epilogue: divide by l, store O^T [HW,C] bf16 ----
  float lrow[8];
  #pragma unroll
  for (int r = 0; r < 8; ++r) lrow[r] = 1.0f / l_run[r + half8];
  #pragma unroll
  for (int t = 0; t < 8; ++t) {
    const int col = c0 + t * 16 + (lane & 15);
    #pragma unroll
    for (int r = 0; r < 8; ++r) {
      const int row = n0 + r + half8;
      Ob[(size_t)row * CH + col] = (bf16_t)(oacc[t][r] * lrow[r]);
    }
  }
}

// ---------------------------------------------------------------------------
// Host orchestration. Workspace (~66.3 MB):
//   stats (4 x B*C f32) | 4 bf16 weight copies | 4 rotating 16 MB bf16 regions
// Region timeline: R1=cnT ->(FT=R4)  R2=snT ->(GT=R1)  R3=sT ->(H=R2)
//                  flash(R4,R1,R2) -> OT=R3 -> final conv + residual -> d_out
// ---------------------------------------------------------------------------
extern "C" void kernel_launch(void* const* d_in, const int* in_sizes, int n_in,
                              void* d_out, int out_size, void* d_ws, size_t ws_size,
                              hipStream_t stream) {
  (void)in_sizes; (void)n_in; (void)out_size; (void)ws_size;
  const float* content = (const float*)d_in[0];
  const float* style   = (const float*)d_in[1];
  const float* Wf = (const float*)d_in[2];
  const float* bf = (const float*)d_in[3];
  const float* Wg = (const float*)d_in[4];
  const float* bg = (const float*)d_in[5];
  const float* Wh = (const float*)d_in[6];
  const float* bh = (const float*)d_in[7];
  const float* Wo = (const float*)d_in[8];
  const float* bo = (const float*)d_in[9];
  float* out = (float*)d_out;

  char* ws = (char*)d_ws;
  size_t off = 0;
  auto carve = [&](size_t bytes) -> char* {
    char* p = ws + off;
    off = (off + bytes + 255) & ~(size_t)255;
    return p;
  };
  const size_t statsB = (size_t)BATCH * CH * sizeof(float);
  const size_t wB     = (size_t)CH * CH * sizeof(bf16_t);
  const size_t regB   = (size_t)BATCH * HWSZ * CH * sizeof(bf16_t);  // 16 MB
  float*  meanC = (float*)carve(statsB);
  float*  rstdC = (float*)carve(statsB);
  float*  meanS = (float*)carve(statsB);
  float*  rstdS = (float*)carve(statsB);
  bf16_t* Wf16  = (bf16_t*)carve(wB);
  bf16_t* Wg16  = (bf16_t*)carve(wB);
  bf16_t* Wh16  = (bf16_t*)carve(wB);
  bf16_t* Wo16  = (bf16_t*)carve(wB);
  bf16_t* R1 = (bf16_t*)carve(regB);
  bf16_t* R2 = (bf16_t*)carve(regB);
  bf16_t* R3 = (bf16_t*)carve(regB);
  bf16_t* R4 = (bf16_t*)carve(regB);

  const long long sHC = (long long)HWSZ * CH;   // per-batch stride, [HW,C]
  const long long sCH = (long long)CH * HWSZ;   // per-batch stride, [C,HW]

  // 1. stats
  stats_kernel<<<dim3(BATCH * CH), 256, 0, stream>>>(content, meanC, rstdC);
  stats_kernel<<<dim3(BATCH * CH), 256, 0, stream>>>(style,   meanS, rstdS);

  // 2. bf16 weight copies
  const int wn = CH * CH;
  f32_to_bf16_kernel<<<(wn + 255) / 256, 256, 0, stream>>>(Wf, Wf16, wn);
  f32_to_bf16_kernel<<<(wn + 255) / 256, 256, 0, stream>>>(Wg, Wg16, wn);
  f32_to_bf16_kernel<<<(wn + 255) / 256, 256, 0, stream>>>(Wh, Wh16, wn);
  f32_to_bf16_kernel<<<(wn + 255) / 256, 256, 0, stream>>>(Wo, Wo16, wn);

  // 3. normalize + transpose to [HW,C] bf16
  dim3 tb(32, 8), tg(HWSZ / 32, CH / 32, BATCH);
  transnorm_kernel<<<tg, tb, 0, stream>>>(content, R1, meanC, rstdC, 1);
  transnorm_kernel<<<tg, tb, 0, stream>>>(style,   R2, meanS, rstdS, 1);
  transnorm_kernel<<<tg, tb, 0, stream>>>(style,   R3, meanS, rstdS, 0);

  // 4. F^T = cnT x Wf^T + bf  -> R4 [HW,C]
  gemm_nt_kernel<1, false, true><<<dim3(CH / 64, HWSZ / 64, BATCH), 256, 0, stream>>>(
      R1, CH, sHC, Wf16, CH, 0, bf, nullptr, 0, R4, CH, sHC, CH);
  // 5. G^T = snT x Wg^T + bg  -> R1 [HW,C]
  gemm_nt_kernel<1, false, true><<<dim3(CH / 64, HWSZ / 64, BATCH), 256, 0, stream>>>(
      R2, CH, sHC, Wg16, CH, 0, bg, nullptr, 0, R1, CH, sHC, CH);
  // 6. H = Wh x style + bh    -> R2 [C,HW]
  gemm_nt_kernel<2, false, true><<<dim3(HWSZ / 64, CH / 64, BATCH), 256, 0, stream>>>(
      Wh16, CH, 0, R3, CH, sHC, bh, nullptr, 0, R2, HWSZ, sCH, CH);

  // 7. fused attention: softmax(F^T G) H^T -> O^T in R3 [HW,C]
  flash_attn_kernel<<<dim3(HWSZ / 16, BATCH), 256, 0, stream>>>(R4, R1, R2, R3);

  // 8. out = Wo x O + bo + content  (f32, [B,C,H,W])
  gemm_nt_kernel<2, true, false><<<dim3(HWSZ / 64, CH / 64, BATCH), 256, 0, stream>>>(
      Wo16, CH, 0, R3, CH, sHC, bo, content, sCH, out, HWSZ, sCH, CH);
}
